// LSTMlayer_12876311953520
// MI455X (gfx1250) — matrix operand order repaired and checked by reading, not compiled, not run on hardware
//
#include <hip/hip_runtime.h>
#include <hip/hip_bf16.h>

typedef __bf16 bf16_t;
typedef __attribute__((ext_vector_type(8)))  bf16_t v8bf;
typedef __attribute__((ext_vector_type(16))) bf16_t v16bf;
typedef __attribute__((ext_vector_type(8)))  float  v8f;

// Exact param types for the async-to-LDS builtin (from compiler diagnostic):
// pointer to "int __attribute__((vector_size(16)))" in global / LDS address spaces.
typedef int v4i_raw __attribute__((__vector_size__(16)));
typedef __attribute__((address_space(1))) v4i_raw* gv4i_p;
typedef __attribute__((address_space(3))) v4i_raw* lv4i_p;

#define T_STEPS 2048
#define ISZ 256
#define HSZ 512
#define BSZ 64
#define NWG 32
#define NTHREADS 128

#define WHH_S 520   // padded K-stride (bf16 elems) for W_hh slice and h^T
#define WIH_S 264   // padded K-stride for W_ih slice and x^T
#define GQ_S  68    // gates row stride (floats), staggers LDS banks

// LDS layout (bytes)
#define OFF_WHH   0
#define OFF_WIH   66560                 // 64*520*2
#define OFF_HT    100352                // + 64*264*2
#define OFF_XT    166912                // + 64*520*2
#define OFF_GATES 200704                // + 64*264*2
#define SMEM_BYTES 218112               // + 64*68*4

#if __has_builtin(__builtin_amdgcn_global_load_async_to_lds_b128)
#define HAVE_ASYNC_LDS 1
#else
#define HAVE_ASYNC_LDS 0
#endif

__device__ __forceinline__ float sigmoidf_(float x) { return 1.0f / (1.0f + __expf(-x)); }

__device__ __forceinline__ v16bf cat16(v8bf lo, v8bf hi) {
  return __builtin_shufflevector(lo, hi, 0,1,2,3,4,5,6,7,8,9,10,11,12,13,14,15);
}

__device__ __forceinline__ void wait_async_zero() {
#if HAVE_ASYNC_LDS
#if __has_builtin(__builtin_amdgcn_s_wait_asynccnt)
  __builtin_amdgcn_s_wait_asynccnt(0);
#else
  asm volatile("s_wait_asynccnt 0" ::: "memory");
#endif
#endif
}

// Monotonic-counter grid barrier; counter zeroed by hipMemsetAsync each launch.
__device__ __forceinline__ void grid_barrier(unsigned int* cnt, unsigned int target) {
  __threadfence();
  __syncthreads();
  if (threadIdx.x == 0) {
    __hip_atomic_fetch_add(cnt, 1u, __ATOMIC_ACQ_REL, __HIP_MEMORY_SCOPE_AGENT);
    while (__hip_atomic_load(cnt, __ATOMIC_ACQUIRE, __HIP_MEMORY_SCOPE_AGENT) < target) {
      __builtin_amdgcn_s_sleep(1);
    }
  }
  __syncthreads();
}

__global__ void __launch_bounds__(NTHREADS)
bnlstm_persistent(const float* __restrict__ x,
                  const float* __restrict__ w_ih,
                  const float* __restrict__ w_hh,
                  const float* __restrict__ bias,
                  const float* __restrict__ g_ih, const float* __restrict__ b_ih,
                  const float* __restrict__ g_hh, const float* __restrict__ b_hh,
                  const float* __restrict__ g_c,  const float* __restrict__ b_c,
                  float* __restrict__ out,
                  bf16_t* __restrict__ hT0,        // ping: h^T [64][512] bf16 (zeroed)
                  bf16_t* __restrict__ hT1,        // pong: h^T [64][512] bf16
                  unsigned int* __restrict__ barcnt)
{
  extern __shared__ char smem[];
  bf16_t* sWhh = (bf16_t*)(smem + OFF_WHH);   // [64 rows][520]
  bf16_t* sWih = (bf16_t*)(smem + OFF_WIH);   // [64 rows][264]
  bf16_t* sHT  = (bf16_t*)(smem + OFF_HT);    // [64 b][520] h transposed
  bf16_t* sXT  = (bf16_t*)(smem + OFF_XT);    // [64 b][264] x_t transposed
  float*  sG   = (float*)(smem + OFF_GATES);  // [4*16 rows][68] normalized gates

  const int tid   = threadIdx.x;
  const int wg    = blockIdx.x;       // 0..31, owns hidden units wg*16..wg*16+15
  const int w     = tid >> 5;         // wave id == gate quarter (i,f,g,o)
  const int lane  = tid & 31;
  const int lhalf = lane >> 4;
  const int l16   = lane & 15;

  // ---- Stage weight slices to LDS as bf16 (once) ----
  for (int e = tid; e < 64 * HSZ; e += NTHREADS) {
    int lr = e >> 9, k = e & 511;
    int grow = (lr >> 4) * HSZ + wg * 16 + (lr & 15);
    sWhh[lr * WHH_S + k] = (bf16_t)w_hh[grow * HSZ + k];
  }
  for (int e = tid; e < 64 * ISZ; e += NTHREADS) {
    int lr = e >> 8, k = e & 255;
    int grow = (lr >> 4) * HSZ + wg * 16 + (lr & 15);
    sWih[lr * WIH_S + k] = (bf16_t)w_ih[grow * ISZ + k];
  }

  // Per-lane BN params for the 8 C-tile rows this lane touches
  float gA[8], bA[8], gB[8], bB[8], bs[8];
  #pragma unroll
  for (int r = 0; r < 8; ++r) {
    int grow = w * HSZ + wg * 16 + r + 8 * lhalf;
    gA[r] = g_hh[grow]; bA[r] = b_hh[grow];
    gB[r] = g_ih[grow]; bB[r] = b_ih[grow];
    bs[r] = bias[grow];
  }

  // c/h-phase constants: thread owns (hidden row cj, batch cb..cb+7)
  const int   cj  = tid >> 3;
  const int   cb  = (tid & 7) << 3;
  const float gcv = g_c[wg * 16 + cj];
  const float bcv = b_c[wg * 16 + cj];
  float creg[8];
  #pragma unroll
  for (int k = 0; k < 8; ++k) creg[k] = 0.0f;

  for (int t = 0; t < T_STEPS; ++t) {
    const bf16_t* hsrc = (t & 1) ? hT1 : hT0;   // read h_t
    bf16_t*       hdst = (t & 1) ? hT0 : hT1;   // write h_{t+1}

    // ---- Snapshot h^T (global bf16) into LDS: 4096 x 16B chunks ----
#if HAVE_ASYNC_LDS
    for (int c = tid; c < 4096; c += NTHREADS) {
      int b = c >> 6, off = (c & 63) << 3;
      __builtin_amdgcn_global_load_async_to_lds_b128(
          (gv4i_p)(hsrc + b * HSZ + off),
          (lv4i_p)(sHT + b * WHH_S + off),
          0, 0);
    }
#else
    for (int c = tid; c < 4096; c += NTHREADS) {
      int b = c >> 6, off = (c & 63) << 3;
      *(uint4*)(sHT + b * WHH_S + off) = *(const uint4*)(hsrc + b * HSZ + off);
    }
#endif
    // ---- Stage x_t transposed ----
    const float* xt = x + (size_t)t * (ISZ * BSZ);
    for (int e = tid; e < ISZ * BSZ; e += NTHREADS) {
      int i = e >> 6, b = e & 63;
      sXT[b * WIH_S + i] = (bf16_t)xt[e];
    }
    // Prefetch next timestep's x tile (64 KB across 128 lanes)
    if (t + 1 < T_STEPS) {
      __builtin_prefetch((const char*)(x + (size_t)(t + 1) * (ISZ * BSZ)) + tid * 512, 0, 0);
    }
    wait_async_zero();
    __syncthreads();

    // ---- WMMA GEMMs: a = W_hh@h , b = W_ih@x_t ; each wave: 16 rows x 64 cols ----
    v8f accA[4] = {};
    v8f accB[4] = {};
    const bf16_t* arowHH = sWhh + (w * 16 + l16) * WHH_S;
    const bf16_t* arowIH = sWih + (w * 16 + l16) * WIH_S;

    #pragma unroll 4
    for (int kt = 0; kt < 16; ++kt) {
      const int k0 = kt * 32;
      const bf16_t* pa = arowHH + k0 + lhalf * 8;
      v16bf af = cat16(*(const v8bf*)pa, *(const v8bf*)(pa + 16));
      #pragma unroll
      for (int n = 0; n < 4; ++n) {
        const bf16_t* pb = sHT + (n * 16 + l16) * WHH_S + k0 + lhalf * 8;
        v16bf bf = cat16(*(const v8bf*)pb, *(const v8bf*)(pb + 16));
        accA[n] = __builtin_amdgcn_wmma_f32_16x16x32_bf16(
            false, af, false, bf, (short)0, accA[n], false, false);
      }
    }
    #pragma unroll 2
    for (int kt = 0; kt < 8; ++kt) {
      const int k0 = kt * 32;
      const bf16_t* pa = arowIH + k0 + lhalf * 8;
      v16bf af = cat16(*(const v8bf*)pa, *(const v8bf*)(pa + 16));
      #pragma unroll
      for (int n = 0; n < 4; ++n) {
        const bf16_t* pb = sXT + (n * 16 + l16) * WIH_S + k0 + lhalf * 8;
        v16bf bf = cat16(*(const v8bf*)pb, *(const v8bf*)(pb + 16));
        accB[n] = __builtin_amdgcn_wmma_f32_16x16x32_bf16(
            false, af, false, bf, (short)0, accB[n], false, false);
      }
    }

    // ---- Row-wise BN over batch (64 cols) directly on C fragments ----
    #pragma unroll
    for (int r = 0; r < 8; ++r) {
      float sa = 0.f, qa = 0.f, sb = 0.f, qb = 0.f;
      #pragma unroll
      for (int n = 0; n < 4; ++n) {
        float va = accA[n][r]; sa += va; qa += va * va;
        float vb = accB[n][r]; sb += vb; qb += vb * vb;
      }
      #pragma unroll
      for (int m = 1; m < 16; m <<= 1) {   // reduce within 16-lane half
        sa += __shfl_xor(sa, m, 32);  qa += __shfl_xor(qa, m, 32);
        sb += __shfl_xor(sb, m, 32);  qb += __shfl_xor(qb, m, 32);
      }
      const float ma = sa * (1.f / 64.f);
      const float mb = sb * (1.f / 64.f);
      const float ia = rsqrtf(qa * (1.f / 64.f) - ma * ma + 1e-5f) * gA[r];
      const float ib = rsqrtf(qb * (1.f / 64.f) - mb * mb + 1e-5f) * gB[r];
      float* grow_p = sG + (w * 16 + (r + 8 * lhalf)) * GQ_S;
      #pragma unroll
      for (int n = 0; n < 4; ++n) {
        grow_p[n * 16 + l16] =
            (accA[n][r] - ma) * ia + bA[r] + (accB[n][r] - mb) * ib + bB[r] + bs[r];
      }
    }
    __syncthreads();

    // ---- Cell & hidden update: c' = sig(f)c + sig(i)tanh(g); h = sig(o)tanh(BN(c')) ----
    float csum = 0.f, csq = 0.f, oreg[8];
    #pragma unroll
    for (int k = 0; k < 8; ++k) {
      const int b = cb + k;
      float xi = sG[(0 * 16 + cj) * GQ_S + b];
      float xf = sG[(1 * 16 + cj) * GQ_S + b];
      float xg = sG[(2 * 16 + cj) * GQ_S + b];
      oreg[k]  = sG[(3 * 16 + cj) * GQ_S + b];
      float c = sigmoidf_(xf) * creg[k] + sigmoidf_(xi) * tanhf(xg);
      creg[k] = c;  csum += c;  csq += c * c;
    }
    #pragma unroll
    for (int m = 1; m < 8; m <<= 1) {       // 8 threads cover one hidden row
      csum += __shfl_xor(csum, m, 32);
      csq  += __shfl_xor(csq,  m, 32);
    }
    const float cm = csum * (1.f / 64.f);
    const float ci = rsqrtf(csq * (1.f / 64.f) - cm * cm + 1e-5f) * gcv;

    float hv[8];
    #pragma unroll
    for (int k = 0; k < 8; ++k) {
      hv[k] = sigmoidf_(oreg[k]) * tanhf((creg[k] - cm) * ci + bcv);
    }
    float* op = out + (size_t)t * (HSZ * BSZ) + (wg * 16 + cj) * BSZ + cb;
    *(float4*)(op)     = make_float4(hv[0], hv[1], hv[2], hv[3]);
    *(float4*)(op + 4) = make_float4(hv[4], hv[5], hv[6], hv[7]);
    #pragma unroll
    for (int k = 0; k < 8; ++k) {
      hdst[(cb + k) * HSZ + wg * 16 + cj] = (bf16_t)hv[k];
    }

    // Single barrier per step (double-buffered h): publish h_{t+1} device-wide.
    grid_barrier(barcnt, (unsigned)(t + 1) * NWG);
  }
}

extern "C" void kernel_launch(void* const* d_in, const int* in_sizes, int n_in,
                              void* d_out, int out_size, void* d_ws, size_t ws_size,
                              hipStream_t stream) {
  (void)in_sizes; (void)n_in; (void)out_size; (void)ws_size;
  const float* x    = (const float*)d_in[0];
  const float* wih  = (const float*)d_in[1];
  const float* whh  = (const float*)d_in[2];
  const float* bias = (const float*)d_in[3];
  const float* gih  = (const float*)d_in[4];
  const float* bih  = (const float*)d_in[5];
  const float* ghh  = (const float*)d_in[6];
  const float* gc   = (const float*)d_in[8];
  const float* bhh  = (const float*)d_in[7];
  const float* bc   = (const float*)d_in[9];
  float*        out = (float*)d_out;
  bf16_t*       hT0 = (bf16_t*)d_ws;                               // 64 KB ping
  bf16_t*       hT1 = (bf16_t*)((char*)d_ws + 64 * 1024);          // 64 KB pong
  unsigned int* cnt = (unsigned int*)((char*)d_ws + 128 * 1024);   // barrier counter

  // h0 = 0 (both buffers) and barrier counter = 0 (graph-capture-safe memset node)
  (void)hipMemsetAsync(d_ws, 0, 128 * 1024 + 256, stream);

  hipLaunchKernelGGL(bnlstm_persistent, dim3(NWG), dim3(NTHREADS), SMEM_BYTES, stream,
                     x, wih, whh, bias, gih, bih, ghh, bhh, gc, bc, out, hT0, hT1, cnt);
}
